// SimCLRLoss_3874060501803
// MI455X (gfx1250) — matrix operand order repaired and checked by reading, not compiled, not run on hardware
//
#include <hip/hip_runtime.h>

// ---------------------------------------------------------------------------
// SimCLR (NT-Xent) loss, fused for MI455X / gfx1250 (wave32, WMMA).
//
//   z  = [l2norm(z_i); l2norm(z_j)]            (8192 x 128, stored f16 in ws)
//   sim = z z^T / 0.5                          (never materialized)
//   lse_i = log( exp(pos_i) + sum_{j != i} exp(sim_ij) )   (pos counted twice)
//   loss  = mean(lse_i - pos_i)
//
// Key points:
//  * v_wmma_f32_16x16x32_f16: 4 issues per 16x16 tile (K=128), f32 accum.
//  * sim/T <= 2 (cosine <= 1) -> fixed logsumexp max of 2.0 => single pass.
//  * 2 MB f16 z panel is fully L2-resident; no HBM re-traffic.
// ---------------------------------------------------------------------------

typedef __attribute__((ext_vector_type(16))) _Float16 v16h;
typedef __attribute__((ext_vector_type(8)))  float    v8f;

#define NROWS  4096
#define D      128
#define M2     (2 * NROWS)          // 8192 total rows
#define INV_T  2.0f                 // 1 / TEMPERATURE
#define MAXLOG 2.0f                 // upper bound of sim / T

// ------------- Phase 1: L2-normalize rows, emit f16 z [8192][128] ----------
__global__ void __launch_bounds__(256)
simclr_normalize(const float* __restrict__ zi,
                 const float* __restrict__ zj,
                 _Float16* __restrict__ zh)
{
    const int wave = (blockIdx.x * blockDim.x + threadIdx.x) >> 5; // 1 wave/row
    const int lane = threadIdx.x & 31;
    if (wave >= M2) return;

    const float* src = (wave < NROWS) ? (zi + (size_t)wave * D)
                                      : (zj + (size_t)(wave - NROWS) * D);
    float4 v = reinterpret_cast<const float4*>(src)[lane];   // 4 floats / lane
    float ss = v.x * v.x + v.y * v.y + v.z * v.z + v.w * v.w;
#pragma unroll
    for (int m = 16; m >= 1; m >>= 1) ss += __shfl_xor(ss, m, 32);
    const float s = 1.0f / fmaxf(sqrtf(ss), 1e-12f);

    union { _Float16 h[4]; uint2 u; } o;
    o.h[0] = (_Float16)(v.x * s);
    o.h[1] = (_Float16)(v.y * s);
    o.h[2] = (_Float16)(v.z * s);
    o.h[3] = (_Float16)(v.w * s);
    reinterpret_cast<uint2*>(zh + (size_t)wave * D)[lane] = o.u;
}

__global__ void simclr_zero(float* out) { out[0] = 0.0f; }

// ------------- Phase 2: fused z z^T + masked logsumexp ---------------------
// One wave per 16-row stripe; loop over 512 column tiles of 16.
__global__ void __launch_bounds__(256)
simclr_lse(const _Float16* __restrict__ zh, float* __restrict__ out)
{
    const int lane = threadIdx.x & 31;
    const int wave = blockIdx.x * 8 + (threadIdx.x >> 5);   // 512 waves total
    const int rowBase = wave * 16;
    const int nlo = lane & 15;
    const int hi  = lane >> 4;

    // ---- A fragments: 16 rows x 128 K of f16 (ISA 16-bit A layout) ----
    // lanes 0-15: K = {kb..kb+7, kb+16..kb+23}; lanes 16-31: +8 on each chunk
    const _Float16* arow = zh + (size_t)(rowBase + nlo) * D;
    v16h A[4];
#pragma unroll
    for (int f = 0; f < 4; ++f) {
        union { uint4 u[2]; v16h h; } t;
        const int kb = f * 32 + hi * 8;
        t.u[0] = *reinterpret_cast<const uint4*>(arow + kb);
        t.u[1] = *reinterpret_cast<const uint4*>(arow + kb + 16);
        A[f] = t.h;
    }

    // Global row index and positive-pair column per C VGPR slot.
    int m[8], p[8];
#pragma unroll
    for (int k = 0; k < 8; ++k) {
        m[k] = rowBase + hi * 8 + k;
        p[k] = (m[k] + NROWS) & (M2 - 1);
    }

    float sm[8], posv[8];
#pragma unroll
    for (int k = 0; k < 8; ++k) { sm[k] = 0.0f; posv[k] = 0.0f; }

    for (int nt = 0; nt < M2 / 16; ++nt) {
        // B = z^T: lane nlo holds column (tile base + nlo) = row of z;
        // lanes 0-15: K = {kb..kb+7, kb+16..kb+23}; lanes 16-31: +8 on each chunk
        const _Float16* brow = zh + (size_t)(nt * 16 + nlo) * D;
        v8f c = {};
#pragma unroll
        for (int f = 0; f < 4; ++f) {
            union { uint4 u[2]; v16h h; } t;
            const int kb = f * 32 + hi * 8;
            t.u[0] = *reinterpret_cast<const uint4*>(brow + kb);
            t.u[1] = *reinterpret_cast<const uint4*>(brow + kb + 16);
            c = __builtin_amdgcn_wmma_f32_16x16x32_f16(
                    false, A[f], false, t.h, (short)0, c, false, false);
        }

        const int col = nt * 16 + nlo;       // this lane's column in the tile
#pragma unroll
        for (int k = 0; k < 8; ++k) {
            const float v = c[k] * INV_T;                    // sim / T
            if (col != m[k]) sm[k] += __expf(v - MAXLOG);    // diag masked
            if (col == p[k]) posv[k] = v;                    // positive logit
        }
    }

    // Reduce exp-sums / positives across the 16 lanes sharing each row.
#pragma unroll
    for (int k = 0; k < 8; ++k) {
#pragma unroll
        for (int msk = 8; msk >= 1; msk >>= 1) {
            sm[k]   += __shfl_xor(sm[k],   msk, 16);
            posv[k] += __shfl_xor(posv[k], msk, 16);  // only one lane nonzero
        }
    }

    if (nlo == 0) {   // lanes 0 and 16 each finalize their 8 rows
        float acc = 0.0f;
#pragma unroll
        for (int k = 0; k < 8; ++k) {
            const float S = sm[k] + __expf(posv[k] - MAXLOG); // prepended pos
            acc += (MAXLOG + __logf(S)) - posv[k];
        }
        atomicAdd(out, acc * (1.0f / (float)M2));
    }
}

// ---------------------------------------------------------------------------
extern "C" void kernel_launch(void* const* d_in, const int* in_sizes, int n_in,
                              void* d_out, int out_size, void* d_ws, size_t ws_size,
                              hipStream_t stream)
{
    const float* zi = (const float*)d_in[0];
    const float* zj = (const float*)d_in[1];
    _Float16* zh = (_Float16*)d_ws;          // 8192*128*2 = 2 MB scratch
    float* out = (float*)d_out;

    simclr_normalize<<<M2 / 8, 256, 0, stream>>>(zi, zj, zh); // 8 rows/block
    simclr_zero<<<1, 1, 0, stream>>>(out);
    simclr_lse<<<(M2 / 16) / 8, 256, 0, stream>>>(zh, out);   // 64 blocks
}
